// PairwiseAttention_10849087390539
// MI455X (gfx1250) — compile-verified, matrix operand order checked by reading, not run on hardware
//
#include <hip/hip_runtime.h>

typedef __attribute__((ext_vector_type(16))) _Float16 v16h;
typedef __attribute__((ext_vector_type(8)))  float    v8f;

#define WMMA_F32_F16(a,b,c) __builtin_amdgcn_wmma_f32_16x16x32_f16(false,(a),false,(b),(short)0,(c),false,false)

// Load a 16x32 f16 A-fragment (also valid for B when given B^T stored [N][K])
// from row-major storage with leading dimension `ld` (in elements).
// ISA 7.12.2: lane 0-15 -> row = lane, lanes 16-31 -> row = lane-16 with K offset +8;
// VGPR j in 0..3 -> K = 2*j (+half*8), VGPR 4..7 -> K = 16 + 2*(j-4) (+half*8).
__device__ __forceinline__ v16h load_frag_rm(const _Float16* __restrict__ p, int ld) {
  const int lane = threadIdx.x & 31;
  const int row  = lane & 15;
  const int half = lane >> 4;
  v16h f;
  #pragma unroll
  for (int j = 0; j < 8; ++j) {
    const int kb = ((j & 3) << 1) + (half << 3) + ((j >> 2) << 4);
    f[2 * j]     = p[row * ld + kb];
    f[2 * j + 1] = p[row * ld + kb + 1];
  }
  return f;
}

// Async global -> LDS copy of 16 bytes per lane (CDNA5 GLOBAL_LOAD_ASYNC_TO_LDS_B128,
// tracked by ASYNCcnt). lds_off = low 32 bits of the generic shared-pointer.
__device__ __forceinline__ void async_copy_b128(unsigned lds_off, unsigned long long gaddr) {
  asm volatile("global_load_async_to_lds_b128 %0, %1, off"
               :: "v"(lds_off), "v"(gaddr) : "memory");
}
__device__ __forceinline__ void wait_async0() {
  asm volatile("s_wait_asynccnt 0" ::: "memory");
}

// ---------------- kernel 0: weights -> f16, transposed to [N][K] -----------
__global__ __launch_bounds__(256) void k_convert(const float* __restrict__ wqkv,
                                                 const float* __restrict__ wproj,
                                                 _Float16* __restrict__ wt,
                                                 _Float16* __restrict__ wpt) {
  const int idx = blockIdx.x * 256 + threadIdx.x;
  if (idx < 384 * 128) {
    const int n = idx >> 7, k = idx & 127;
    wt[n * 128 + k] = (_Float16)wqkv[k * 384 + n];
  }
  if (idx < 128 * 128) {
    const int n = idx >> 7, k = idx & 127;
    wpt[n * 128 + k] = (_Float16)wproj[k * 128 + n];
  }
}

// ---------------- kernel 1: LayerNorm + QKV projection ---------------------
// grid = 4096 (16-token M-tiles), block = 128 (4 waves, 6 N-tiles each of 24)
__global__ __launch_bounds__(128) void k_ln_qkv(const float* __restrict__ z,
                                                const float* __restrict__ ln_g,
                                                const float* __restrict__ ln_b,
                                                const _Float16* __restrict__ wt,
                                                const float* __restrict__ b_qkv,
                                                _Float16* __restrict__ q,
                                                _Float16* __restrict__ k,
                                                _Float16* __restrict__ vt) {
  __shared__ _Float16 zn[16][128];
  const int tid  = threadIdx.x;
  const int tok0 = blockIdx.x * 16;

  { // LayerNorm: 8 lanes per token, 16 features per lane
    const int t = tid >> 3;
    const int l = tid & 7;
    const float* zp = z + (size_t)(tok0 + t) * 128 + l * 16;
    float vbuf[16];
    float s = 0.f, ss = 0.f;
    #pragma unroll
    for (int i = 0; i < 16; ++i) { float x = zp[i]; vbuf[i] = x; s += x; ss += x * x; }
    #pragma unroll
    for (int off = 1; off < 8; off <<= 1) {
      s  += __shfl_xor(s,  off, 32);
      ss += __shfl_xor(ss, off, 32);
    }
    const float mu   = s * (1.f / 128.f);
    const float var  = ss * (1.f / 128.f) - mu * mu;
    const float rstd = rsqrtf(var + 1e-5f);
    #pragma unroll
    for (int i = 0; i < 16; ++i) {
      const int d = l * 16 + i;
      zn[t][d] = (_Float16)((vbuf[i] - mu) * rstd * ln_g[d] + ln_b[d]);
    }
  }
  __syncthreads();

  const int wave = tid >> 5;
  v8f acc[6] = {};
  #pragma unroll
  for (int kc = 0; kc < 4; ++kc) {
    const v16h a = load_frag_rm(&zn[0][0] + kc * 32, 128);
    #pragma unroll
    for (int j = 0; j < 6; ++j) {
      const int nt = wave * 6 + j;
      const v16h b = load_frag_rm(wt + nt * 16 * 128 + kc * 32, 128);
      acc[j] = WMMA_F32_F16(a, b, acc[j]);
    }
  }

  // scatter q / k / v(T) as f16
  const int lane  = tid & 31;
  const int n     = lane & 15;
  const int halfw = lane >> 4;
  #pragma unroll
  for (int j = 0; j < 6; ++j) {
    const int f     = (wave * 6 + j) * 16 + n;     // 0..383
    const int which = f >> 7;                      // 0=q 1=k 2=v (tiles never straddle)
    const int dall  = f & 127;
    const int h     = dall >> 5, d = dall & 31;
    const float bias = b_qkv[f];
    #pragma unroll
    for (int i = 0; i < 8; ++i) {
      const int tok = tok0 + halfw * 8 + i;        // M = i + 8*half
      const int r = tok >> 8, c = tok & 255;
      const _Float16 val = (_Float16)(acc[j][i] + bias);
      const size_t rh = (size_t)(r * 4 + h);
      if (which == 0)      q [(rh * 256 + c) * 32 + d] = val;
      else if (which == 1) k [(rh * 256 + c) * 32 + d] = val;
      else                 vt[(rh * 32 + d) * 256 + c] = val;   // transposed
    }
  }
}

// ---------------- kernel 2: per-(row,head) attention ------------------------
// grid = 1024 (r*4+h), block = 128 (4 waves); each wave: 4 sequential 16-query tiles
__global__ __launch_bounds__(128) void k_attn(const _Float16* __restrict__ q,
                                              const _Float16* __restrict__ kk,
                                              const _Float16* __restrict__ vt,
                                              _Float16* __restrict__ o) {
  __shared__ _Float16 ksm[256 * 32];        // 16 KB: K tile [key][d]
  __shared__ _Float16 vsm[32 * 256];        // 16 KB: V^T tile [d][key]
  __shared__ _Float16 psm[4][16 * 256];     // 32 KB: per-wave softmax'd P
  const int tid = threadIdx.x;
  const int rh  = blockIdx.x;
  const int r = rh >> 2, h = rh & 3;

  { // async global->LDS staging of K and V^T (2048 x b128, 16 per thread)
    const unsigned long long kg = (unsigned long long)(uintptr_t)(kk + (size_t)rh * 8192);
    const unsigned long long vg = (unsigned long long)(uintptr_t)(vt + (size_t)rh * 8192);
    const unsigned kl = (unsigned)(uintptr_t)&ksm[0];
    const unsigned vl = (unsigned)(uintptr_t)&vsm[0];
    #pragma unroll
    for (int i = 0; i < 8; ++i) {
      const unsigned b = (unsigned)(tid + i * 128) * 16u;   // byte offset 0..16K
      async_copy_b128(kl + b, kg + b);
      async_copy_b128(vl + b, vg + b);
    }
    wait_async0();
  }
  __syncthreads();

  const int wave = tid >> 5, lane = tid & 31;
  const int n = lane & 15, halfw = lane >> 4;
  const float scale = 0.17677669529663687f;  // 1/sqrt(32)
  const _Float16* qbase = q + (size_t)rh * 8192;
  _Float16* pw = &psm[wave][0];

  for (int mt = 0; mt < 4; ++mt) {
    const int q0 = (wave * 4 + mt) * 16;
    const v16h a = load_frag_rm(qbase + q0 * 32, 32);   // K-dim=32: one WMMA per tile
    v8f s[16];
    #pragma unroll
    for (int nt = 0; nt < 16; ++nt) {
      const v16h b = load_frag_rm(ksm + nt * 16 * 32, 32);
      v8f cz = {};
      s[nt] = WMMA_F32_F16(a, b, cz);
    }
    // softmax over 256 keys; row m = i + 8*half lives in one 16-lane half-wave
    float mx[8], rs[8];
    #pragma unroll
    for (int i = 0; i < 8; ++i) {
      float m = -1e30f;
      #pragma unroll
      for (int nt = 0; nt < 16; ++nt) m = fmaxf(m, s[nt][i]);
      #pragma unroll
      for (int off = 1; off < 16; off <<= 1) m = fmaxf(m, __shfl_xor(m, off, 32));
      mx[i] = m * scale;
    }
    #pragma unroll
    for (int nt = 0; nt < 16; ++nt)
      #pragma unroll
      for (int i = 0; i < 8; ++i)
        s[nt][i] = __expf(s[nt][i] * scale - mx[i]);
    #pragma unroll
    for (int i = 0; i < 8; ++i) {
      float sm = 0.f;
      #pragma unroll
      for (int nt = 0; nt < 16; ++nt) sm += s[nt][i];
      #pragma unroll
      for (int off = 1; off < 16; off <<= 1) sm += __shfl_xor(sm, off, 32);
      rs[i] = 1.f / sm;
    }
    #pragma unroll
    for (int nt = 0; nt < 16; ++nt)
      #pragma unroll
      for (int i = 0; i < 8; ++i)
        pw[(halfw * 8 + i) * 256 + nt * 16 + n] = (_Float16)(s[nt][i] * rs[i]);
    asm volatile("s_wait_dscnt 0" ::: "memory");   // wave-private LDS RAW

    // O = P(16x256) @ V(256x32); both operands from LDS
    v8f oc[2] = {};
    #pragma unroll
    for (int kt = 0; kt < 8; ++kt) {
      const v16h pa = load_frag_rm(pw + kt * 32, 256);
      #pragma unroll
      for (int nt2 = 0; nt2 < 2; ++nt2) {
        const v16h b = load_frag_rm(vsm + nt2 * 16 * 256 + kt * 32, 256);
        oc[nt2] = WMMA_F32_F16(pa, b, oc[nt2]);
      }
    }
    #pragma unroll
    for (int nt2 = 0; nt2 < 2; ++nt2)
      #pragma unroll
      for (int i = 0; i < 8; ++i) {
        const int qi = q0 + halfw * 8 + i;
        o[((size_t)(r * 256 + qi)) * 128 + h * 32 + nt2 * 16 + n] = (_Float16)oc[nt2][i];
      }
  }
}

// ---------------- kernel 3: output projection + bias + residual ------------
// grid = 1024, block = 128 (4 waves, one 16-token tile per wave)
__global__ __launch_bounds__(128) void k_proj(const _Float16* __restrict__ o,
                                              const _Float16* __restrict__ wpt,
                                              const float* __restrict__ b_proj,
                                              const float* __restrict__ z,
                                              float* __restrict__ out) {
  const int tid = threadIdx.x;
  const int wave = tid >> 5, lane = tid & 31;
  const int n = lane & 15, halfw = lane >> 4;
  const int mt = blockIdx.x * 4 + wave;
  const size_t tok0 = (size_t)mt * 16;

  v8f acc[8] = {};
  #pragma unroll
  for (int kc = 0; kc < 4; ++kc) {
    const v16h a = load_frag_rm(o + tok0 * 128 + kc * 32, 128);
    #pragma unroll
    for (int nt = 0; nt < 8; ++nt) {
      const v16h b = load_frag_rm(wpt + nt * 16 * 128 + kc * 32, 128);
      acc[nt] = WMMA_F32_F16(a, b, acc[nt]);
    }
  }
  #pragma unroll
  for (int nt = 0; nt < 8; ++nt) {
    const int f = nt * 16 + n;
    const float bias = b_proj[f];
    #pragma unroll
    for (int i = 0; i < 8; ++i) {
      const size_t idx = (tok0 + halfw * 8 + i) * 128 + f;
      out[idx] = z[idx] + acc[nt][i] + bias;
    }
  }
}

extern "C" void kernel_launch(void* const* d_in, const int* in_sizes, int n_in,
                              void* d_out, int out_size, void* d_ws, size_t ws_size,
                              hipStream_t stream) {
  const float* z      = (const float*)d_in[0];
  const float* ln_g   = (const float*)d_in[1];
  const float* ln_b   = (const float*)d_in[2];
  const float* w_qkv  = (const float*)d_in[3];
  const float* b_qkv  = (const float*)d_in[4];
  const float* w_proj = (const float*)d_in[5];
  const float* b_proj = (const float*)d_in[6];
  float* out = (float*)d_out;

  _Float16* ws   = (_Float16*)d_ws;
  _Float16* wt   = ws;                      // 384*128
  _Float16* wpt  = wt  + 49152;             // 128*128
  _Float16* qws  = wpt + 16384;             // 256*4*256*32
  _Float16* kws  = qws + 8388608;
  _Float16* vtws = kws + 8388608;           // transposed V
  _Float16* ows  = vtws + 8388608;          // attention output, f16

  k_convert<<<192, 256, 0, stream>>>(w_qkv, w_proj, wt, wpt);
  k_ln_qkv <<<4096, 128, 0, stream>>>(z, ln_g, ln_b, wt, b_qkv, qws, kws, vtws);
  k_attn   <<<1024, 128, 0, stream>>>(qws, kws, vtws, ows);
  k_proj   <<<1024, 128, 0, stream>>>(ows, wpt, b_proj, z, out);
}
